// RWKVTimeMix_2018634629180
// MI455X (gfx1250) — matrix operand order repaired and checked by reading, not compiled, not run on hardware
//
#include <hip/hip_runtime.h>

// ---------------------------------------------------------------------------
// RWKV time-mix for MI455X (gfx1250, wave32, WMMA + async-to-LDS).
//
//   0) split_bf16  : fp32 -> (hi, lo) bf16 planes for input, w_rvk, w_out
//   1) gemm        : rvk[M,3C] = input @ w_rvk^T      (bf16x3 WMMA, fp32 out)
//   2) wkv_scan    : serial-in-T WKV recurrence; emits sigmoid(r)*y directly
//                    as bf16 hi/lo planes (reusing the input-plane scratch),
//                    final (a,b,p) states into d_out's h1/c sections.
//   3) gemm        : out[M,C] = rwkv @ w_out^T
//
// GEMM design: 128x128x32 tiles, 8 waves, wave = 64x32 (4x2 WMMA tiles),
// 24 x v_wmma_f32_16x16x32_bf16 per K-step (hi*hi + hi*lo + lo*hi ~ fp32).
// Tiles stream into LDS with global_load_async_to_lds_b128 (ASYNCcnt),
// double-buffered (2 stages x 4 planes x 8KB = 64KB LDS), stored pre-swizzled
// in WMMA fragment order so a fragment is one contiguous 32B LDS read.
// Problem is HBM-bound (~1.8GB traffic -> ~75us at 23.3 TB/s); 1.65 TFLOP of
// bf16 WMMA is noise against that, so fp32-accurate bf16x3 is free.
// ---------------------------------------------------------------------------

typedef __attribute__((ext_vector_type(16))) __bf16 v16bf;
typedef __attribute__((ext_vector_type(8)))  float  v8f;
typedef __attribute__((ext_vector_type(4)))  float  f4v;
typedef __attribute__((ext_vector_type(4)))  unsigned short us4;

#define BM 128
#define BN 128
#define BK 32

union Frag { v16bf v; unsigned u[8]; };

__device__ __forceinline__ unsigned short bf16_rne(float x) {
  unsigned u = __float_as_uint(x);
  u += 0x7FFFu + ((u >> 16) & 1u);            // round-to-nearest-even
  return (unsigned short)(u >> 16);
}
__device__ __forceinline__ float bf16_to_f32(unsigned short s) {
  return __uint_as_float(((unsigned)s) << 16);
}

// gfx1250 async copy: 16B global -> LDS, tracked by ASYNCcnt (no VGPR round-trip)
__device__ __forceinline__ void async_copy16(unsigned lds_off, const void* gptr) {
  asm volatile("global_load_async_to_lds_b128 %0, %1, off"
               :: "v"(lds_off), "v"(gptr) : "memory");
}
__device__ __forceinline__ void wait_async0() {
  asm volatile("s_wait_asynccnt 0x0" ::: "memory");
}

// --------------------------- fp32 -> bf16 hi/lo split ----------------------
__global__ __launch_bounds__(256)
void split_bf16(const float* __restrict__ x, unsigned short* __restrict__ hi,
                unsigned short* __restrict__ lo, long n) {
  long i = ((long)blockIdx.x * blockDim.x + threadIdx.x) * 4;
  if (i >= n) return;
  f4v v = *(const f4v*)(x + i);
  us4 h, l;
#pragma unroll
  for (int e = 0; e < 4; ++e) {
    h[e] = bf16_rne(v[e]);
    l[e] = bf16_rne(v[e] - bf16_to_f32(h[e]));
  }
  *(us4*)(hi + i) = h;
  *(us4*)(lo + i) = l;
}

// ------------------------------- GEMM --------------------------------------
// C[M,N] = (Ahi+Alo)[M,K] * (Whi+Wlo)[N,K]^T  (bf16 planes, fp32 out)
__global__ __launch_bounds__(256)
void gemm_bf16x3_async(const unsigned short* __restrict__ Ahi,
                       const unsigned short* __restrict__ Alo, long lda,
                       const unsigned short* __restrict__ Bhi,
                       const unsigned short* __restrict__ Blo, long ldb,
                       float* __restrict__ Cg, long ldc, int K) {
  // [stage][plane][4096] : plane 0=Ahi 1=Alo 2=Bhi 3=Blo ; 64KB total
  __shared__ unsigned short smem[2][4][4096];

  const int tid  = threadIdx.x;
  const int lane = tid & 31;
  const int wave = tid >> 5;
  const int wm   = wave >> 2;           // 0..1  -> 64-row slab
  const int wn   = wave & 3;            // 0..3  -> 32-col slab
  const int half = lane >> 4;
  const int l15  = lane & 15;
  const long m0  = (long)blockIdx.y * BM;
  const long n0  = (long)blockIdx.x * BN;

  const unsigned ldsA  = (unsigned)(size_t)&smem[0][0][0]; // truncate -> LDS byte offset
  const unsigned PLANE = 4096u * 2u;                       // 8KB
  const unsigned STAGE = 4u * PLANE;                       // 32KB

  // Issue one stage worth of async copies: 512 16B chunks per plane, 4 planes,
  // 256 threads -> 8 async b128 per thread. LDS destinations are pre-swizzled
  // into WMMA fragment order (lane-contiguous 32B rows).
  auto load_stage = [&](int stage, int kt) {
    const unsigned base = ldsA + (unsigned)stage * STAGE;
#pragma unroll
    for (int it = 0; it < 2; ++it) {
      int ch = tid + it * 256;          // 0..511
      int r  = ch >> 2;                 // tile row 0..127
      int c0 = (ch & 3) << 3;           // k-chunk base: 0,8,16,24
      // A-fragment swizzle: subtile s=r>>4, lane = ((c0>>3)&1)*16 + (r&15),
      // byte-half selected by c0>>4
      unsigned aoff = (unsigned)(((((r >> 4) << 5) + (((c0 >> 3) & 1) << 4) + (r & 15)) << 5)
                                 + ((c0 >> 4) << 4));
      // B-fragment swizzle: lane = ((c0>>3)&1)*16 + (r&15), byte-half by c0>>4
      unsigned boff = (unsigned)(((((r >> 4) << 5) + (((c0 >> 3) & 1) << 4) + (r & 15)) << 5)
                                 + ((c0 >> 4) << 4));
      async_copy16(base + 0 * PLANE + aoff, Ahi + (m0 + r) * lda + kt + c0);
      async_copy16(base + 1 * PLANE + aoff, Alo + (m0 + r) * lda + kt + c0);
      async_copy16(base + 2 * PLANE + boff, Bhi + (n0 + r) * ldb + kt + c0);
      async_copy16(base + 3 * PLANE + boff, Blo + (n0 + r) * ldb + kt + c0);
    }
  };

  v8f acc[4][2];
#pragma unroll
  for (int i = 0; i < 4; ++i)
#pragma unroll
    for (int j = 0; j < 2; ++j) acc[i][j] = 0;

  load_stage(0, 0);

  for (int kt = 0, st = 0; kt < K; kt += BK, st ^= 1) {
    wait_async0();          // this wave's async copies for stage `st` landed
    __syncthreads();        // everyone's copies landed; prev readers done
    if (kt + BK < K) load_stage(st ^ 1, kt + BK);

    const unsigned short* sa_hi = &smem[st][0][0];
    const unsigned short* sa_lo = &smem[st][1][0];
    const unsigned short* sb_hi = &smem[st][2][0];
    const unsigned short* sb_lo = &smem[st][3][0];

    Frag ahi[4], alo[4], bhi[2], blo[2];
#pragma unroll
    for (int i = 0; i < 4; ++i) {
      int s = wm * 4 + i;
      ahi[i].v = *(const v16bf*)(sa_hi + ((s << 5) + lane) * 16);  // 32B/lane
      alo[i].v = *(const v16bf*)(sa_lo + ((s << 5) + lane) * 16);
    }
#pragma unroll
    for (int j = 0; j < 2; ++j) {
      int s = wn * 2 + j;
      bhi[j].v = *(const v16bf*)(sb_hi + ((s << 5) + lane) * 16);
      blo[j].v = *(const v16bf*)(sb_lo + ((s << 5) + lane) * 16);
    }

#pragma unroll
    for (int i = 0; i < 4; ++i)
#pragma unroll
      for (int j = 0; j < 2; ++j) {
        acc[i][j] = __builtin_amdgcn_wmma_f32_16x16x32_bf16(
            false, ahi[i].v, false, bhi[j].v, (short)0, acc[i][j], false, false);
        acc[i][j] = __builtin_amdgcn_wmma_f32_16x16x32_bf16(
            false, ahi[i].v, false, blo[j].v, (short)0, acc[i][j], false, false);
        acc[i][j] = __builtin_amdgcn_wmma_f32_16x16x32_bf16(
            false, alo[i].v, false, bhi[j].v, (short)0, acc[i][j], false, false);
      }
    __syncthreads();        // all waves done reading stage `st`
  }

  // epilogue: C/D layout -> M = vgpr + 8*half, N = lane%16
#pragma unroll
  for (int i = 0; i < 4; ++i)
#pragma unroll
    for (int j = 0; j < 2; ++j) {
      long mg = m0 + wm * 64 + i * 16 + half * 8;
      long ng = n0 + wn * 32 + j * 16 + l15;
#pragma unroll
      for (int v = 0; v < 8; ++v)
        Cg[(mg + v) * ldc + ng] = acc[i][j][v];
    }
}

// ------------------------------- WKV scan ----------------------------------
// One lane per (b,c); serial over T. Emits sigmoid(r)*y as bf16 hi/lo planes.
__global__ __launch_bounds__(256)
void wkv_scan(const float* __restrict__ rvk,          // (B*T, 3C) fp32
              unsigned short* __restrict__ rwkv_hi,   // (B*T, C) bf16
              unsigned short* __restrict__ rwkv_lo,   // (B*T, C) bf16
              const float* __restrict__ hidden,       // (B, C)
              const float* __restrict__ time_decay,
              const float* __restrict__ time_first,
              float* __restrict__ h1_out,             // (B, C)
              float* __restrict__ c_out,              // (B, 2C)
              int Bn, int T, int C) {
  int idx = blockIdx.x * blockDim.x + threadIdx.x;
  if (idx >= Bn * C) return;
  int b = idx / C, c = idx % C;
  const float w = -__expf(time_decay[c]);
  const float u = time_first[c];
  float a  = hidden[(size_t)b * C + c];
  float bb = 0.f;
  float p  = -1e38f;
  const int C3 = 3 * C;
  const float* base = rvk + (size_t)b * T * C3;
  for (int t = 0; t < T; ++t) {
    const float* row = base + (size_t)t * C3;
    float rr = row[c];
    float vv = row[C + c];
    float kk = row[2 * C + c];
    // output using carried state
    float ww = u + kk;
    float q  = fmaxf(p, ww);
    float e1 = __expf(p - q);
    float e2 = __expf(ww - q);
    float y  = fmaf(e1, a, e2 * vv) / fmaf(e1, bb, e2);
    float ry = y / (1.f + __expf(-rr));               // sigmoid(r) * wkv
    size_t o = ((size_t)b * T + t) * C + c;
    unsigned short h = bf16_rne(ry);
    rwkv_hi[o] = h;
    rwkv_lo[o] = bf16_rne(ry - bf16_to_f32(h));
    // state update
    float ww2 = p + w;
    float q2  = fmaxf(ww2, kk);
    float e1b = __expf(ww2 - q2);
    float e2b = __expf(kk - q2);
    a  = fmaf(e1b, a, e2b * vv);
    bb = fmaf(e1b, bb, e2b);
    p  = q2;
  }
  h1_out[(size_t)b * C + c]        = a;
  c_out[(size_t)b * 2 * C + c]     = bb;
  c_out[(size_t)b * 2 * C + C + c] = p;
}

// ------------------------------- host --------------------------------------
extern "C" void kernel_launch(void* const* d_in, const int* in_sizes, int n_in,
                              void* d_out, int out_size, void* d_ws, size_t ws_size,
                              hipStream_t stream) {
  const float* input      = (const float*)d_in[0];
  const float* hidden     = (const float*)d_in[1];
  const float* w_rvk      = (const float*)d_in[2];
  const float* w_out      = (const float*)d_in[3];
  const float* time_decay = (const float*)d_in[4];
  const float* time_first = (const float*)d_in[5];

  const int  C  = in_sizes[4];                  // 2048
  const int  Bn = in_sizes[1] / C;              // 16
  const int  T  = in_sizes[0] / (Bn * C);       // 1024
  const int  C3 = 3 * C;
  const long M  = (long)Bn * T;                 // 16384

  // workspace carve-up (bytes): rvk fp32 | A hi/lo planes (reused as rwkv) |
  // w_rvk hi/lo | w_out hi/lo  -> ~604 MB total
  char* ws = (char*)d_ws;
  float*          rvk    = (float*)ws;                    ws += (size_t)M * C3 * 4;
  unsigned short* aHi    = (unsigned short*)ws;           ws += (size_t)M * C * 2;
  unsigned short* aLo    = (unsigned short*)ws;           ws += (size_t)M * C * 2;
  unsigned short* wrvkHi = (unsigned short*)ws;           ws += (size_t)C3 * C * 2;
  unsigned short* wrvkLo = (unsigned short*)ws;           ws += (size_t)C3 * C * 2;
  unsigned short* woutHi = (unsigned short*)ws;           ws += (size_t)C * C * 2;
  unsigned short* woutLo = (unsigned short*)ws;

  float* out = (float*)d_out;                   // (B,T,C)
  float* h1  = out + (size_t)M * C;             // (B,C)
  float* cc  = h1 + (size_t)Bn * C;             // (B,2C)

  // 0) precision-split passes
  {
    long nIn = (long)M * C, nWr = (long)C3 * C, nWo = (long)C * C;
    split_bf16<<<dim3((unsigned)((nIn / 4 + 255) / 256)), 256, 0, stream>>>(input, aHi, aLo, nIn);
    split_bf16<<<dim3((unsigned)((nWr / 4 + 255) / 256)), 256, 0, stream>>>(w_rvk, wrvkHi, wrvkLo, nWr);
    split_bf16<<<dim3((unsigned)((nWo / 4 + 255) / 256)), 256, 0, stream>>>(w_out, woutHi, woutLo, nWo);
  }
  // 1) rvk = input @ w_rvk^T
  {
    dim3 grid(C3 / BN, (unsigned)(M / BM));
    gemm_bf16x3_async<<<grid, 256, 0, stream>>>(aHi, aLo, (long)C,
                                                wrvkHi, wrvkLo, (long)C,
                                                rvk, (long)C3, C);
  }
  // 2) WKV scan + fused sigmoid(r)*y -> rwkv planes (reuse aHi/aLo scratch)
  {
    int total = Bn * C;
    wkv_scan<<<(total + 255) / 256, 256, 0, stream>>>(rvk, aHi, aLo, hidden,
                                                      time_decay, time_first,
                                                      h1, cc, Bn, T, C);
  }
  // 3) out = rwkv @ w_out^T
  {
    dim3 grid(C / BN, (unsigned)(M / BM));
    gemm_bf16x3_async<<<grid, 256, 0, stream>>>(aHi, aLo, (long)C,
                                                woutHi, woutLo, (long)C,
                                                out, (long)C, C);
  }
}
